// ConLoss_3427383902250
// MI455X (gfx1250) — matrix operand order repaired and checked by reading, not compiled, not run on hardware
//
#include <hip/hip_runtime.h>
#include <hip/hip_bf16.h>

typedef __attribute__((ext_vector_type(16))) _Float16 v16h;
typedef __attribute__((ext_vector_type(8)))  _Float16 h8;
typedef __attribute__((ext_vector_type(4)))  _Float16 h4;
typedef __attribute__((ext_vector_type(8)))  float    v8f;

union V16 { v16h v; h8 h[2]; };

#define NROWS_TOT 8192
#define D         512
#define HALF_B    4096
#define PITCH     520            // 512 + 8 halves: b128 rows land on distinct bank groups
#define BLK_ROWS  64
#define COL_CHUNK 1024
#define NCHUNK    8
#define NTILES    64             // COL_CHUNK / 16
#define INV_T     2.0f           // 1 / 0.5

// ---- CDNA5 async helpers ---------------------------------------------------
__device__ __forceinline__ void wait_asynccnt_le8() {
#if __has_builtin(__builtin_amdgcn_s_wait_asynccnt)
    __builtin_amdgcn_s_wait_asynccnt(8);
#else
    asm volatile("s_wait_asynccnt 0x8" ::: "memory");
#endif
}
__device__ __forceinline__ void wait_asynccnt_0() {
#if __has_builtin(__builtin_amdgcn_s_wait_asynccnt)
    __builtin_amdgcn_s_wait_asynccnt(0);
#else
    asm volatile("s_wait_asynccnt 0x0" ::: "memory");
#endif
}

// Issue one 16-column tile (16 zn rows x 512 halves) as 8 async b128 loads per
// thread: hardware DMA global -> LDS, tracked by ASYNCcnt.
__device__ __forceinline__ void async_copy_tile(const _Float16* __restrict__ zn,
                                                int j0, uint32_t ldsDstBase, int t) {
#pragma unroll
    for (int i = 0; i < 8; ++i) {
        const int idx = t + 128 * i;          // 1024 16B chunks, 128 threads
        const int r = idx >> 6, c = idx & 63; // 64 chunks per row
        const uint32_t ldsa = ldsDstBase + (uint32_t)(r * PITCH + c * 8) * 2u;
        const uint64_t ga =
            (uint64_t)(uintptr_t)(zn + (size_t)(j0 + r) * D + c * 8);
        asm volatile("global_load_async_to_lds_b128 %0, %1, off"
                     :: "v"(ldsa), "v"(ga) : "memory");
    }
}

// ---------------------------------------------------------------- kernel A
// Normalize rows of z = [aug1; aug2] with eps clamp, store as f16.
__global__ __launch_bounds__(256) void norm_rows(const float* __restrict__ a1,
                                                 const float* __restrict__ a2,
                                                 _Float16* __restrict__ zn) {
    const int row  = blockIdx.x * 8 + (threadIdx.x >> 5);   // 8 waves/block, 1 row/wave
    const int lane = threadIdx.x & 31;
    const float* __restrict__ src =
        (row < HALF_B) ? (a1 + (size_t)row * D) : (a2 + (size_t)(row - HALF_B) * D);

    float4 v[4];
    float ss = 0.f;
#pragma unroll
    for (int i = 0; i < 4; ++i) {
        v[i] = ((const float4*)src)[lane + 32 * i];          // cols 4*(lane+32i)..+3
        ss += v[i].x * v[i].x + v[i].y * v[i].y + v[i].z * v[i].z + v[i].w * v[i].w;
    }
#pragma unroll
    for (int off = 16; off >= 1; off >>= 1) ss += __shfl_xor(ss, off, 32);

    const float inv = 1.0f / fmaxf(sqrtf(ss), 1e-8f);
    _Float16* __restrict__ dst = zn + (size_t)row * D;
#pragma unroll
    for (int i = 0; i < 4; ++i) {
        h4 h;
        h.x = (_Float16)(v[i].x * inv);
        h.y = (_Float16)(v[i].y * inv);
        h.z = (_Float16)(v[i].z * inv);
        h.w = (_Float16)(v[i].w * inv);
        ((h4*)dst)[lane + 32 * i] = h;
    }
}

// ---------------------------------------------------------------- kernel B
// Fused sim-tile GEMM (WMMA f16->f32) + online logsumexp over a column chunk.
// grid = (8192/64, 8); block = 128 (4 waves, each wave owns 16 rows).
// B fragments (the wave's rows, full K) live in registers for the whole kernel;
// the 16-column A tiles are double-buffered in LDS and filled by async DMA.
__global__ __launch_bounds__(128) void simtile(const _Float16* __restrict__ zn,
                                               float* __restrict__ pM,
                                               float* __restrict__ pS,
                                               float* __restrict__ pP) {
    extern __shared__ _Float16 lds[];
    _Float16* Rs = lds;                      // [BLK_ROWS][PITCH] block's rows (staging)
    // Cs double buffer: lds + (BLK_ROWS + 16*buf)*PITCH, each [16][PITCH]
    const uint32_t lds_base = (uint32_t)(uintptr_t)lds;   // low 32 bits = LDS offset

    const int t      = threadIdx.x;
    const int lane   = t & 31;
    const int wave   = t >> 5;               // 0..3
    const int nloc   = lane & 15;            // row-local index inside wave's 16 rows
    const int hi     = lane >> 4;            // K/col half select
    const int row0   = blockIdx.x * BLK_ROWS;
    const int j0base = blockIdx.y * COL_CHUNK;
    const int grow   = row0 + wave * 16 + nloc;

    // Stage the block's 64 rows into LDS (coalesced, synchronous; done once).
    for (int idx = t; idx < BLK_ROWS * (D / 8); idx += 128) {
        const int r = idx >> 6, c = idx & 63;
        *(h8*)&Rs[r * PITCH + c * 8] = *(const h8*)&zn[(size_t)(row0 + r) * D + c * 8];
    }
    // Kick off the first column tile into buffer 0 while rows land.
    async_copy_tile(zn, j0base, lds_base + (uint32_t)(BLK_ROWS * PITCH) * 2u, t);
    __syncthreads();                         // Rs visible block-wide

    // Hoist this wave's B fragments (16 rows x full K) into registers.
    // B 32x16 f16 as loaded here (contiguous k order, A uses the same): lane -> N = lane%16; 16 halves -> K = 16*(lane/16)+0..15
    h8 bLo[16], bHi[16];
    {
        const _Float16* rowp = &Rs[(wave * 16 + nloc) * PITCH];
#pragma unroll
        for (int kc = 0; kc < 16; ++kc) {
            bLo[kc] = *(const h8*)&rowp[kc * 32 + 16 * hi];
            bHi[kc] = *(const h8*)&rowp[kc * 32 + 16 * hi + 8];
        }
    }

    const float NEG_INF = -__builtin_inff();
    float m = NEG_INF, s = 0.f, pos = NEG_INF;

    for (int jt = 0; jt < NTILES; ++jt) {
        const int j0 = j0base + jt * 16;
        // Prefetch next tile into the other buffer via async DMA.
        if (jt + 1 < NTILES) {
            const uint32_t nb =
                lds_base + (uint32_t)((BLK_ROWS + 16 * ((jt + 1) & 1)) * PITCH) * 2u;
            async_copy_tile(zn, j0 + 16, nb, t);
            wait_asynccnt_le8();             // current tile done (in-order completion)
        } else {
            wait_asynccnt_0();
        }
        __syncthreads();                     // tile jt complete block-wide

        const _Float16* Cb = lds + (BLK_ROWS + 16 * (jt & 1)) * PITCH;

        // D[c][n] = sum_k Cs[c][k] * Rs[wave*16+n][k]  (A = cols, B = rows-in-regs)
        v8f acc = {0.f, 0.f, 0.f, 0.f, 0.f, 0.f, 0.f, 0.f};
#pragma unroll
        for (int kc = 0; kc < 16; ++kc) {
            V16 a, b;
            // A 16x32 f16: lane -> M=nloc; 16 halves -> K=16*hi+0..15
            a.h[0] = *(const h8*)&Cb[nloc * PITCH + kc * 32 + 16 * hi];
            a.h[1] = *(const h8*)&Cb[nloc * PITCH + kc * 32 + 16 * hi + 8];
            b.h[0] = bLo[kc];
            b.h[1] = bHi[kc];
            acc = __builtin_amdgcn_wmma_f32_16x16x32_f16(
                false, a.v, false, b.v, (short)0, acc, false, false);
        }

        // Online logsumexp: this lane's 8 values all belong to row `grow`,
        // columns j0 + r + 8*hi.
#pragma unroll
        for (int r = 0; r < 8; ++r) {
            const int gcol = j0 + r + 8 * hi;
            const float v = INV_T * acc[r];
            if (gcol == grow + HALF_B || gcol + HALF_B == grow) pos = v;
            if (gcol != grow) {
                if (v <= m) {
                    s += __expf(v - m);
                } else {
                    s = s * __expf(m - v) + 1.0f;
                    m = v;
                }
            }
        }
        __syncthreads();                     // buffer free before next async overwrite
    }

    // Merge the two lanes (L, L+16) that share this row.
    const float mo = __shfl_xor(m, 16, 32);
    const float so = __shfl_xor(s, 16, 32);
    const float po = __shfl_xor(pos, 16, 32);
    const float M  = fmaxf(m, mo);
    const float S  = s * __expf(m - M) + so * __expf(mo - M);
    const float P  = fmaxf(pos, po);
    if (hi == 0) {
        const size_t o = (size_t)blockIdx.y * NROWS_TOT + grow;
        pM[o] = M; pS[o] = S; pP[o] = P;
    }
}

// ---------------------------------------------------------------- kernel C
__global__ __launch_bounds__(256) void merge_rows(const float* __restrict__ pM,
                                                  const float* __restrict__ pS,
                                                  const float* __restrict__ pP,
                                                  float* __restrict__ loss) {
    const int i = blockIdx.x * 256 + threadIdx.x;
    float M = -__builtin_inff();
#pragma unroll
    for (int c = 0; c < NCHUNK; ++c) M = fmaxf(M, pM[(size_t)c * NROWS_TOT + i]);
    float S = 0.f, P = -__builtin_inff();
#pragma unroll
    for (int c = 0; c < NCHUNK; ++c) {
        S += pS[(size_t)c * NROWS_TOT + i] * __expf(pM[(size_t)c * NROWS_TOT + i] - M);
        P = fmaxf(P, pP[(size_t)c * NROWS_TOT + i]);
    }
    loss[i] = (M + __logf(S)) - P;      // -pos/T + logsumexp(sim/T)
}

// ---------------------------------------------------------------- kernel D
__global__ __launch_bounds__(256) void final_reduce(const float* __restrict__ loss,
                                                    float* __restrict__ out) {
    __shared__ float red[256];
    const int t = threadIdx.x;
    float a = 0.f;
    for (int i = t; i < NROWS_TOT; i += 256) a += loss[i];   // fixed order
    red[t] = a;
    __syncthreads();
    for (int w = 128; w > 0; w >>= 1) {
        if (t < w) red[t] += red[t + w];
        __syncthreads();
    }
    if (t == 0) out[0] = red[0] / (float)NROWS_TOT;
}

// ---------------------------------------------------------------- launch
extern "C" void kernel_launch(void* const* d_in, const int* in_sizes, int n_in,
                              void* d_out, int out_size, void* d_ws, size_t ws_size,
                              hipStream_t stream) {
    const float* a1 = (const float*)d_in[0];
    const float* a2 = (const float*)d_in[1];
    float* out = (float*)d_out;

    char* ws = (char*)d_ws;
    _Float16* zn = (_Float16*)ws;                                 // 8 MB
    float* pM   = (float*)(ws + (size_t)NROWS_TOT * D * 2);       // 8*8192 f32
    float* pS   = pM + (size_t)NCHUNK * NROWS_TOT;
    float* pP   = pS + (size_t)NCHUNK * NROWS_TOT;
    float* loss = pP + (size_t)NCHUNK * NROWS_TOT;                // 8192 f32

    norm_rows<<<NROWS_TOT / 8, 256, 0, stream>>>(a1, a2, zn);

    dim3 grid(NROWS_TOT / BLK_ROWS, NCHUNK);                      // 128 x 8
    size_t shmem = (size_t)(BLK_ROWS + 32) * PITCH * sizeof(_Float16); // 99,840 B
    simtile<<<grid, 128, shmem, stream>>>(zn, pM, pS, pP);

    merge_rows<<<NROWS_TOT / 256, 256, 0, stream>>>(pM, pS, pP, loss);
    final_reduce<<<1, 256, 0, stream>>>(loss, out);
}
